// FullEmbedding_44418551775448
// MI455X (gfx1250) — matrix operand order repaired and checked by reading, not compiled, not run on hardware
//
#include <hip/hip_runtime.h>
#include <hip/hip_bf16.h>

typedef __attribute__((ext_vector_type(16))) _Float16 v16h;
typedef __attribute__((ext_vector_type(8)))  _Float16 v8h;
typedef __attribute__((ext_vector_type(4)))  _Float16 v4h;
typedef __attribute__((ext_vector_type(8)))  float    v8f;

#define BS_SL_N   32768   // 32*1024 sequences
#define TSTEPS    16
#define DIN       64
#define HID       128
#define GATES     512     // 4*HID
#define ROWS      32      // sequences per block (two WMMA M tiles)

// ---- WMMA fragment loaders (CDNA5 16-bit layouts, wave32) ----
// A (16x32 f16): lane L -> M = L%16; element e -> K = (e/8)*16 + (L/16)*8 + e%8
__device__ __forceinline__ v16h load_a_frag(const _Float16* base, int row_stride,
                                            int kbase, int lane) {
    const int m    = lane & 15;
    const int half = lane >> 4;
    const _Float16* p = base + m * row_stride + kbase + half * 8;
    union { v16h v; v8h h[2]; } u;
    u.h[0] = *(const v8h*)(p);        // K = kbase + half*8 + 0..7
    u.h[1] = *(const v8h*)(p + 16);   // K = kbase + 16 + half*8 + 0..7
    return u.v;
}

// B (32x16 f16): lane L -> N = n0 + L%16; element e -> K = kbase + (e/8)*16 + (L/16)*8 + e%8
// Weights stored n-major ([N][ldk], i.e. original W row-major) => K contiguous.
__device__ __forceinline__ v16h load_b_frag(const _Float16* w, int ldk,
                                            int n0, int kbase, int lane) {
    const int n    = n0 + (lane & 15);
    const int half = lane >> 4;
    const _Float16* p = w + n * ldk + kbase + half * 8;
    union { v16h v; v8h h[2]; } u;
    u.h[0] = *(const v8h*)(p);        // K = kbase + half*8 + 0..7
    u.h[1] = *(const v8h*)(p + 16);   // K = kbase + 16 + half*8 + 0..7
    return u.v;
}

__device__ __forceinline__ float sigmoidf_fast(float x) {
    return 1.0f / (1.0f + __expf(-x));
}

// Vectorized f32->f16 staging: global float4 loads, packed 8-byte LDS stores.
__device__ __forceinline__ void stage_f16(_Float16* dst, const float* __restrict__ src,
                                          int nelem, int tid) {
    const float4* s4 = (const float4*)src;
    for (int i = tid; i < nelem / 4; i += 256) {
        float4 v = s4[i];
        v4h h = { (_Float16)v.x, (_Float16)v.y, (_Float16)v.z, (_Float16)v.w };
        *(v4h*)(dst + 4 * i) = h;
    }
}

__global__ __launch_bounds__(256, 1)
void lstm_wmma_kernel(const float* __restrict__ xw,   // [N, T, DIN]
                      const int*   __restrict__ xn,   // [N, 4] (last col = length)
                      const float* __restrict__ Wih,  // [GATES, DIN]
                      const float* __restrict__ Whh,  // [GATES, HID]
                      const float* __restrict__ bih,  // [GATES]
                      const float* __restrict__ bhh,  // [GATES]
                      float*       __restrict__ out)  // [N, HID]
{
    __shared__ _Float16 sWih[GATES * DIN];          //  64 KB
    __shared__ _Float16 sWhh[GATES * HID];          // 128 KB
    __shared__ _Float16 sX[ROWS * TSTEPS * DIN];    //  64 KB  [row][t][k]
    __shared__ _Float16 sH[2][ROWS * HID];          //  16 KB  ping-pong h buffers
    __shared__ float    sBias[GATES];               //   2 KB

    const int tid  = threadIdx.x;
    const int lane = tid & 31;
    const int wave = tid >> 5;                      // 8 waves
    const int nbase = blockIdx.x * ROWS;

    // Stage weights + x as f16 (vectorized: global b128 loads, LDS b64 stores)
    stage_f16(sWih, Wih, GATES * DIN, tid);
    stage_f16(sWhh, Whh, GATES * HID, tid);
    stage_f16(sX, xw + (size_t)nbase * (TSTEPS * DIN), ROWS * TSTEPS * DIN, tid);
    for (int i = tid; i < GATES; i += 256) sBias[i] = bih[i] + bhh[i];
    // h0 = 0 (only buffer 0 is read at t=0)
    for (int i = tid; i < ROWS * HID; i += 256) sH[0][i] = (_Float16)0.0f;
    __syncthreads();

    // Wave 'wave' owns hidden columns [wave*16, wave*16+16) for ALL 4 gates,
    // for both M tiles (rows 0..15 and 16..31). i/f/g/o for a given (row,col)
    // therefore live in the same lane's accumulators -> cell math on registers.
    const int half = lane >> 4;
    const int mofs = 8 * half;                      // acc element r -> row mofs+r
    const int col  = wave * 16 + (lane & 15);       // hidden column (lane-fixed)

    const float bias_i = sBias[col];
    const float bias_f = sBias[HID + col];
    const float bias_g = sBias[2 * HID + col];
    const float bias_o = sBias[3 * HID + col];

    int tcapA[8], tcapB[8];
#pragma unroll
    for (int r = 0; r < 8; ++r) {
        int lA = xn[(nbase + mofs + r) * 4 + 3];
        int lB = xn[(nbase + 16 + mofs + r) * 4 + 3];
        tcapA[r] = min(max(lA - 1, 0), TSTEPS - 1);
        tcapB[r] = min(max(lB - 1, 0), TSTEPS - 1);
    }

    float cA[8], cB[8], hcapA[8], hcapB[8];
#pragma unroll
    for (int r = 0; r < 8; ++r) { cA[r] = cB[r] = 0.0f; hcapA[r] = hcapB[r] = 0.0f; }

#pragma unroll 1
    for (int t = 0; t < TSTEPS; ++t) {
        const _Float16* hcur = sH[t & 1];
        _Float16*       hnxt = sH[(t & 1) ^ 1];

        v8f accA[4], accB[4];
#pragma unroll
        for (int g = 0; g < 4; ++g) { accA[g] = (v8f){}; accB[g] = (v8f){}; }

        // 6 K-steps: kb 0..1 input (K=64 of Wih), kb 2..5 recurrent (K=128 of Whh).
        // A fragments loaded ONCE per k-step, shared by all 4 gate tiles;
        // B fragments shared by both M tiles.
#pragma unroll
        for (int kb = 0; kb < 6; ++kb) {
            v16h aA, aB;
            if (kb < 2) {
                aA = load_a_frag(sX + t * DIN,                     TSTEPS * DIN, kb * 32, lane);
                aB = load_a_frag(sX + 16 * TSTEPS * DIN + t * DIN, TSTEPS * DIN, kb * 32, lane);
            } else {
                aA = load_a_frag(hcur,            HID, (kb - 2) * 32, lane);
                aB = load_a_frag(hcur + 16 * HID, HID, (kb - 2) * 32, lane);
            }
#pragma unroll
            for (int g = 0; g < 4; ++g) {
                v16h b;
                if (kb < 2) b = load_b_frag(sWih, DIN, g * HID + wave * 16, kb * 32, lane);
                else        b = load_b_frag(sWhh, HID, g * HID + wave * 16, (kb - 2) * 32, lane);
                accA[g] = __builtin_amdgcn_wmma_f32_16x16x32_f16(
                            false, aA, false, b, (short)0, accA[g], false, false);
                accB[g] = __builtin_amdgcn_wmma_f32_16x16x32_f16(
                            false, aB, false, b, (short)0, accB[g], false, false);
            }
        }

        // ---- LSTM cell entirely in registers; h_t goes to the OTHER buffer,
        // so no barrier is needed before writing (other waves may still be
        // reading hcur, never hnxt).
#pragma unroll
        for (int r = 0; r < 8; ++r) {
            {
                float iv = sigmoidf_fast(accA[0][r] + bias_i);
                float fv = sigmoidf_fast(accA[1][r] + bias_f);
                float gv = tanhf(accA[2][r] + bias_g);
                float ov = sigmoidf_fast(accA[3][r] + bias_o);
                cA[r] = fv * cA[r] + iv * gv;
                float h = ov * tanhf(cA[r]);
                if (t == tcapA[r]) hcapA[r] = h;
                hnxt[(mofs + r) * HID + col] = (_Float16)h;
            }
            {
                float iv = sigmoidf_fast(accB[0][r] + bias_i);
                float fv = sigmoidf_fast(accB[1][r] + bias_f);
                float gv = tanhf(accB[2][r] + bias_g);
                float ov = sigmoidf_fast(accB[3][r] + bias_o);
                cB[r] = fv * cB[r] + iv * gv;
                float h = ov * tanhf(cB[r]);
                if (t == tcapB[r]) hcapB[r] = h;
                hnxt[(16 + mofs + r) * HID + col] = (_Float16)h;
            }
        }
        __syncthreads();   // single barrier per step: h_t complete before next GEMM
    }

    // Write captured h_final (f32 output)
#pragma unroll
    for (int r = 0; r < 8; ++r) {
        out[(size_t)(nbase + mofs + r) * HID + col]      = hcapA[r];
        out[(size_t)(nbase + 16 + mofs + r) * HID + col] = hcapB[r];
    }
}

extern "C" void kernel_launch(void* const* d_in, const int* in_sizes, int n_in,
                              void* d_out, int out_size, void* d_ws, size_t ws_size,
                              hipStream_t stream) {
    const float* xw  = (const float*)d_in[0];
    const int*   xn  = (const int*)  d_in[1];
    const float* Wih = (const float*)d_in[2];
    const float* Whh = (const float*)d_in[3];
    const float* bih = (const float*)d_in[4];
    const float* bhh = (const float*)d_in[5];
    float* out = (float*)d_out;

    const int nblocks = BS_SL_N / ROWS;   // 1024
    lstm_wmma_kernel<<<nblocks, 256, 0, stream>>>(xw, xn, Wih, Whh, bih, bhh, out);
}
